// Attention_34840774705279
// MI455X (gfx1250) — matrix operand order repaired and checked by reading, not compiled, not run on hardware
//
#include <hip/hip_runtime.h>

typedef _Float16 h16;
typedef __attribute__((ext_vector_type(16))) _Float16 v16h;
typedef __attribute__((ext_vector_type(8)))  float    v8f;

#define B_    4
#define NTOK  16384
#define DIMC  256
#define NHEAD 8
#define NPTS  4
#define HD_   32
#define HMAP  128
#define WMAP  128
#define KD    256
#define MTOT  (B_*NTOK)

// LDS B tile: 64 columns x 256 K as f16.  Row = 256 halves = 32 uint4,
// padded to 33 uint4 (528 B) so per-column stride hits distinct banks.
#define BROW_Q 33

static __device__ __forceinline__ int clampi(int v, int lo, int hi) {
    return v < lo ? lo : (v > hi ? hi : v);
}

// ---------------------------------------------------------------------------
// f32 -> f16 pre-conversion (8 elements per thread, n % 8 == 0).
// ---------------------------------------------------------------------------
__global__ __launch_bounds__(256)
void cvt_f32_f16(const float* __restrict__ s, h16* __restrict__ d, int n)
{
    const int i = (blockIdx.x * 256 + threadIdx.x) * 8;
    if (i < n) {
        float4 f0 = *(const float4*)(s + i);
        float4 f1 = *(const float4*)(s + i + 4);
        union { h16 h[8]; uint4 u; } pk;
        pk.h[0]=(h16)f0.x; pk.h[1]=(h16)f0.y; pk.h[2]=(h16)f0.z; pk.h[3]=(h16)f0.w;
        pk.h[4]=(h16)f1.x; pk.h[5]=(h16)f1.y; pk.h[6]=(h16)f1.z; pk.h[7]=(h16)f1.w;
        *(uint4*)(d + i) = pk.u;
    }
}

// ---------------------------------------------------------------------------
// WMMA GEMM: out[M, NC] = X[M,256] @ Wh^T + bias  (f16 inputs, f32 accum,
// OutT output).  Bias rows [0,split) from bias0, rest from bias1.
// Block = 256 thr = 8 waves; block tile = 256(M) x 64(N).  B tile async-DMA'd
// into LDS (pure f16 byte copy) once per block; each wave owns two 16-row A
// stripes and 8 accumulators (32 rows x 64 cols).  K = 256 in steps of 32.
// ---------------------------------------------------------------------------
template <typename OutT>
__global__ __launch_bounds__(256)
void gemm_wmma_f16(const h16* __restrict__ X,
                   const h16* __restrict__ Wh,      // [NCW, 256] f16
                   const float* __restrict__ bias0,
                   const float* __restrict__ bias1,
                   int split, int NC,
                   OutT* __restrict__ out)
{
    __shared__ uint4 ldsB[64 * BROW_Q];          // 33,792 bytes

    const int wid  = threadIdx.x >> 5;
    const int lane = threadIdx.x & 31;
    const int half = lane >> 4;
    const int l16  = lane & 15;
    const int nTiles  = (NC + 63) >> 6;
    const int bm      = blockIdx.x / nTiles;
    const int bn      = blockIdx.x % nTiles;
    const int rowBase = bm * 256 + wid * 32;
    const int colBase = bn * 64;

    // ---- Stage B tile -> LDS via async global->LDS DMA (ASYNCcnt path).
    // 4 threads per column, 64 K (=128 B) each: 8 x b128 transfers/thread.
    {
        const int srow = threadIdx.x >> 2;        // 0..63 : column of output
        const int kseg = (threadIdx.x & 3) * 64;  // K segment start (halves)
        int n = colBase + srow;
        if (n > NC - 1) n = NC - 1;               // clamp; stores are guarded
        const uint4* gp = (const uint4*)(Wh + (size_t)n * KD + kseg);
        uint32_t ldsOff =
            (uint32_t)(uintptr_t)&ldsB[srow * BROW_Q + (kseg >> 3)];
#pragma unroll
        for (int q = 0; q < 8; ++q) {
            asm volatile("global_load_async_to_lds_b128 %0, %1, off"
                         :: "v"(ldsOff + 16u * q), "v"(gp + q)
                         : "memory");
        }
        asm volatile("s_wait_asynccnt 0x0" ::: "memory");
    }
    __syncthreads();

    v8f acc0[4] = {};
    v8f acc1[4] = {};

    const h16* __restrict__ arow0 = X + (size_t)(rowBase + l16) * KD;
    const h16* __restrict__ arow1 = arow0 + (size_t)16 * KD;

    for (int kk = 0; kk < KD; kk += 32) {
        // ---- Two A fragments (16x32 f16): lane(l16)=row; e 0..7 ->
        // k=kk+8*half+e, e 8..15 -> k=kk+16+8*half+(e-8).  2 x b128 each.
        const h16* ap0 = arow0 + kk + 8 * half;
        const h16* ap1 = arow1 + kk + 8 * half;
        union { uint4 q[2]; v16h v; } au0, au1;
        au0.q[0] = *(const uint4*)(ap0);
        au0.q[1] = *(const uint4*)(ap0 + 16);
        au1.q[0] = *(const uint4*)(ap1);
        au1.q[1] = *(const uint4*)(ap1 + 16);

        // ---- B fragments from LDS: lane(l16)=col n; e 0..7 -> k=kk+8*half+e, e 8..15 -> k=kk+16+8*half+(e-8).
#pragma unroll
        for (int j = 0; j < 4; ++j) {
            union { uint4 q[2]; v16h v; } bu;
            const int bi = (j * 16 + l16) * BROW_Q + (kk >> 3) + half;
            bu.q[0] = ldsB[bi];
            bu.q[1] = ldsB[bi + 2];
            acc0[j] = __builtin_amdgcn_wmma_f32_16x16x32_f16(
                          false, au0.v, false, bu.v, (short)0, acc0[j], false, false);
            acc1[j] = __builtin_amdgcn_wmma_f32_16x16x32_f16(
                          false, au1.v, false, bu.v, (short)0, acc1[j], false, false);
        }
    }

    // ---- Epilogue.  C/D layout: VGPR r, lane -> (M = r + 8*half, N = l16).
#pragma unroll
    for (int j = 0; j < 4; ++j) {
        const int c = colBase + j * 16 + l16;
        if (c < NC) {
            const float bv = (c < split) ? bias0[c] : bias1[c - split];
#pragma unroll
            for (int r = 0; r < 8; ++r) {
                const int mm = rowBase + r + 8 * half;
                out[(size_t)mm * NC + c]        = (OutT)(acc0[j][r] + bv);
                out[(size_t)(mm + 16) * NC + c] = (OutT)(acc1[j][r] + bv);
            }
        }
    }
}

// ---------------------------------------------------------------------------
// Bilinear point-sampling.  One wave per (g = b*NH + h, n) pair; lane =
// channel within head (HD == 32 == wave32).  ow[M,96]: cols 0..63 offsets
// (incl. bias), 64..95 attn logits (incl. bias).  4-way softmax, zero-padded
// bilinear gather from f16 v map [B,N,256], f32 accum, f16 coalesced store.
// ---------------------------------------------------------------------------
__global__ __launch_bounds__(256)
void sample_points(const h16* __restrict__ vmap,
                   const float* __restrict__ ow,
                   h16* __restrict__ outm)
{
    const int wid  = threadIdx.x >> 5;
    const int lane = threadIdx.x & 31;
    const unsigned pair = blockIdx.x * 8u + (unsigned)wid;  // 0 .. 32*16384-1
    const int g = pair >> 14;          // NTOK = 2^14
    const int n = pair & (NTOK - 1);
    const int b = g >> 3;              // NHEAD = 8
    const int h = g & 7;

    const float rx = (float)(n & (WMAP - 1));   // col
    const float ry = (float)(n >> 7);           // row

    const float* owp = ow + (size_t)(b * NTOK + n) * 96;

    // softmax over the 4 point logits (redundant per lane; values broadcast)
    const float l0 = owp[64 + h * 4 + 0];
    const float l1 = owp[64 + h * 4 + 1];
    const float l2 = owp[64 + h * 4 + 2];
    const float l3 = owp[64 + h * 4 + 3];
    const float mx = fmaxf(fmaxf(l0, l1), fmaxf(l2, l3));
    const float e0 = __expf(l0 - mx), e1 = __expf(l1 - mx);
    const float e2 = __expf(l2 - mx), e3 = __expf(l3 - mx);
    const float inv = 1.0f / (e0 + e1 + e2 + e3);
    const float wsm[4] = { e0 * inv, e1 * inv, e2 * inv, e3 * inv };

    const h16* vbase = vmap + (size_t)b * NTOK * DIMC + h * HD_ + lane;

    const auto corner = [&](int xi, int yi, float wgt) -> float {
        const bool valid = (xi >= 0) & (xi < WMAP) & (yi >= 0) & (yi < HMAP);
        const int xc = clampi(xi, 0, WMAP - 1);
        const int yc = clampi(yi, 0, HMAP - 1);
        const float val = (float)vbase[(size_t)(yc * WMAP + xc) * DIMC];
        return valid ? wgt * val : 0.0f;
    };

    float accv = 0.0f;
#pragma unroll
    for (int p = 0; p < NPTS; ++p) {
        const int oi = (h * NPTS + p) * 2;
        const float gx = rx + owp[oi];       // _grad_scale is identity in fwd
        const float gy = ry + owp[oi + 1];
        const float x0 = floorf(gx), y0 = floorf(gy);
        const float wx1 = gx - x0, wx0 = 1.0f - wx1;
        const float wy1 = gy - y0, wy0 = 1.0f - wy1;
        const int x0i = (int)x0, y0i = (int)y0;
        float s = corner(x0i,     y0i,     wx0 * wy0)
                + corner(x0i + 1, y0i,     wx1 * wy0)
                + corner(x0i,     y0i + 1, wx0 * wy1)
                + corner(x0i + 1, y0i + 1, wx1 * wy1);
        accv += wsm[p] * s;
    }

    outm[(size_t)(b * NTOK + n) * DIMC + h * HD_ + lane] = (h16)accv;
}

// ---------------------------------------------------------------------------
extern "C" void kernel_launch(void* const* d_in, const int* in_sizes, int n_in,
                              void* d_out, int out_size, void* d_ws, size_t ws_size,
                              hipStream_t stream) {
    const float* x      = (const float*)d_in[0];
    const float* v_w    = (const float*)d_in[1];
    const float* v_b    = (const float*)d_in[2];
    const float* aw_w   = (const float*)d_in[3];
    const float* aw_b   = (const float*)d_in[4];
    const float* off_w  = (const float*)d_in[5];
    const float* off_b  = (const float*)d_in[6];
    const float* proj_w = (const float*)d_in[7];
    const float* proj_b = (const float*)d_in[8];
    float* out = (float*)d_out;

    // Workspace layout (all 16B-aligned):
    h16*  xh    = (h16*)d_ws;                       // MTOT*256        (32 MB)
    h16*  wh_v  = xh    + (size_t)MTOT * KD;        // 256*256
    h16*  wh_ow = wh_v  + 256 * KD;                 // 96*256 (off rows 0..63, aw 64..95)
    h16*  wh_p  = wh_ow + 96 * KD;                  // 256*256
    h16*  vh    = wh_p  + 256 * KD;                 // MTOT*256 f16    (32 MB)
    h16*  attnh = vh    + (size_t)MTOT * DIMC;      // MTOT*256 f16    (32 MB)
    float* ow_ws = (float*)(attnh + (size_t)MTOT * DIMC);  // MTOT*96  (25 MB)

    const dim3 blk(256);

    // ---- f32 -> f16 pre-conversion of activations and weights
    cvt_f32_f16<<<dim3((MTOT * KD / 8) / 256), blk, 0, stream>>>(x, xh, MTOT * KD);
    cvt_f32_f16<<<dim3((256 * KD / 8 + 255) / 256), blk, 0, stream>>>(v_w, wh_v, 256 * KD);
    cvt_f32_f16<<<dim3((64 * KD / 8 + 255) / 256), blk, 0, stream>>>(off_w, wh_ow, 64 * KD);
    cvt_f32_f16<<<dim3((32 * KD / 8 + 255) / 256), blk, 0, stream>>>(aw_w, wh_ow + 64 * KD, 32 * KD);
    cvt_f32_f16<<<dim3((256 * KD / 8 + 255) / 256), blk, 0, stream>>>(proj_w, wh_p, 256 * KD);

    // v = x @ v_w^T + v_b   -> f16 [B, N, 256]
    gemm_wmma_f16<h16><<<dim3((MTOT / 256) * (DIMC / 64)), blk, 0, stream>>>(
        xh, wh_v, v_b, v_b, DIMC, DIMC, vh);

    // [offsets | attn logits] = x @ [off_w; aw_w]^T + [off_b; aw_b] -> f32 [M, 96]
    gemm_wmma_f16<float><<<dim3((MTOT / 256) * 2), blk, 0, stream>>>(
        xh, wh_ow, off_b, aw_b, 64, 96, ow_ws);

    // softmax + bilinear point gather -> f16 [B, N, 256]
    sample_points<<<dim3((NHEAD * B_ * NTOK) / 8), blk, 0, stream>>>(
        vh, ow_ws, attnh);

    // out = attn @ proj_w^T + proj_b -> f32
    gemm_wmma_f16<float><<<dim3((MTOT / 256) * (DIMC / 64)), blk, 0, stream>>>(
        attnh, wh_p, proj_b, proj_b, DIMC, DIMC, out);
}
